// Qwen3MoeSparseMoeBlock_31275951849915
// MI455X (gfx1250) — matrix operand order repaired and checked by reading, not compiled, not run on hardware
//
#include <hip/hip_runtime.h>
#include <cstddef>

#define TOKENS 8192
#define HID    2048
#define NEXP   64
#define INTER  768
#define TOPK   8

#define MTILE       64
#define KC          512
#define XS_STRIDE   520          // 512 + 8 bf16 pad (bank-conflict stagger)
#define HS_STRIDE   776          // 768 + 8 bf16 pad
#define FFN_THREADS 512
#define XS_BYTES    (64 * XS_STRIDE * 2)                 // 66560
#define GS_BYTES    (64 * HS_STRIDE * 2)                 // 99328
#define SMEM_BYTES  (XS_BYTES + GS_BYTES + 64 * 4 + 64 * 4)   // 166400

typedef __bf16        v16bf __attribute__((ext_vector_type(16)));
typedef float         v8f   __attribute__((ext_vector_type(8)));
typedef unsigned int  u32x4 __attribute__((ext_vector_type(4)));

// ---------------------------------------------------------------- WMMA helper
__device__ __forceinline__ v8f wmma_bf16(v16bf a, v16bf b, v8f c) {
    // 8 args: (neg_a, A, neg_b, B, c_mod, C, reuse_a, reuse_b)
    return __builtin_amdgcn_wmma_f32_16x16x32_bf16(false, a, false, b, (short)0, c,
                                                   false, false);
}

// A fragment (16x32 bf16) from LDS: elems 0..15 = K[k..k+15]
__device__ __forceinline__ v16bf load_a_lds(const __bf16* p) {
    v16bf a;
    *((u32x4*)&a)       = *(const u32x4*)p;
    *(((u32x4*)&a) + 1) = *(const u32x4*)(p + 8);
    return a;
}

// B fragment (32x16) from row-major f32 weights W[K][ld]. All 16 loads issued
// first (single clause, full MLP), then converted f32->bf16.
__device__ __forceinline__ v16bf load_b_f32(const float* __restrict__ W,
                                            int n, int ck, int ld) {
    const float* p = W + (size_t)ck * ld + n;
    __builtin_prefetch(p + (size_t)32 * ld, 0, 0);   // next K-step into GL2
    float f[16];
#pragma unroll
    for (int j = 0; j < 16; ++j)
        f[j] = p[(size_t)j * ld];
    v16bf b;
#pragma unroll
    for (int j = 0; j < 16; ++j)
        b[j] = (__bf16)f[j];
    return b;
}

// ---------------------------------------------------------------- kernel 0
__global__ __launch_bounds__(256)
void prep_kernel(const float* __restrict__ x, __bf16* __restrict__ xb,
                 float* __restrict__ out, int* __restrict__ ecount, long n) {
    long i = (long)blockIdx.x * blockDim.x + threadIdx.x;
    long stride = (long)gridDim.x * blockDim.x;
    for (; i < n; i += stride) {
        out[i] = 0.0f;
        xb[i]  = (__bf16)x[i];
    }
    if (blockIdx.x == 0 && threadIdx.x < NEXP) ecount[threadIdx.x] = 0;
}

// ---------------------------------------------------------------- kernel 1: router
__global__ __launch_bounds__(256)
void router_kernel(const float* __restrict__ x, const float* __restrict__ gw,
                   int* __restrict__ ecount, int* __restrict__ etok,
                   float* __restrict__ ewt) {
    const int lane = threadIdx.x & 31;
    const int t    = blockIdx.x * 8 + (threadIdx.x >> 5);
    const float* xr = x + (size_t)t * HID;

    float a0 = 0.0f, a1 = 0.0f;          // experts: lane, lane+32
    for (int h = 0; h < HID; ++h) {
        float xv = xr[h];
        a0 = fmaf(xv, gw[h * NEXP + lane],      a0);
        a1 = fmaf(xv, gw[h * NEXP + 32 + lane], a1);
    }

    float mx = fmaxf(a0, a1);
#pragma unroll
    for (int o = 16; o > 0; o >>= 1) mx = fmaxf(mx, __shfl_xor(mx, o, 32));
    float p0 = __expf(a0 - mx), p1 = __expf(a1 - mx);
    float s = p0 + p1;
#pragma unroll
    for (int o = 16; o > 0; o >>= 1) s += __shfl_xor(s, o, 32);
    p0 /= s;  p1 /= s;

    float myw = 0.0f;  int mye = 0;  float wsum = 0.0f;
    for (int k = 0; k < TOPK; ++k) {
        float v;  int ei;
        if (p0 >= p1) { v = p0; ei = lane; } else { v = p1; ei = lane + 32; }
#pragma unroll
        for (int o = 16; o > 0; o >>= 1) {        // argmax, lowest index on tie
            float ov = __shfl_xor(v, o, 32);
            int   oi = __shfl_xor(ei, o, 32);
            if (ov > v || (ov == v && oi < ei)) { v = ov; ei = oi; }
        }
        wsum += v;
        if (lane == k) { myw = v; mye = ei; }
        if (ei < 32) { if (lane == ei)      p0 = -1e30f; }
        else         { if (lane == ei - 32) p1 = -1e30f; }
    }

    if (lane < TOPK) {
        int slot = atomicAdd(&ecount[mye], 1);
        etok[(size_t)mye * TOKENS + slot] = t;
        ewt [(size_t)mye * TOKENS + slot] = myw / wsum;   // norm_topk_prob
    }
}

// X[64 gathered tokens][HID] @ W[HID][INTER] -> acc[4 m-tiles][3 n-tiles] per wave.
// Re-stages the X chunk in LDS each call (cheap vs. weight stream); 12 live v8f
// accumulators keep VGPR pressure low so the 16-load B clauses stay batched.
__device__ __forceinline__ void gemm_xw(const __bf16* __restrict__ xb,
                                        const float* __restrict__ W,
                                        __bf16* Xs, const int* tokS,
                                        int tid, int lane, int wv,
                                        v8f (&acc)[4][3]) {
    const int m15  = lane & 15;
    const int ko   = (lane >> 4) << 3;
    const int kb16 = (lane >> 4) << 4;
    for (int kc = 0; kc < HID; kc += KC) {
        __syncthreads();                            // previous chunk consumed
        {   // stage 64 x KC bf16 activation chunk (vector b128 copies)
            int row  = tid >> 3;
            int part = tid & 7;
            const u32x4* src = (const u32x4*)(xb + (size_t)tokS[row] * HID + kc);
            u32x4*       dst = (u32x4*)(Xs + row * XS_STRIDE);
#pragma unroll
            for (int j = 0; j < 8; ++j) dst[part + j * 8] = src[part + j * 8];
        }
        __syncthreads();

        for (int kk = 0; kk < KC; kk += 32) {
            v16bf af[4];
#pragma unroll
            for (int m = 0; m < 4; ++m)
                af[m] = load_a_lds(Xs + (m * 16 + m15) * XS_STRIDE + kb16 + kk);
            const int ck = kc + kk + kb16;
#pragma unroll
            for (int nl = 0; nl < 3; ++nl) {
                v16bf b = load_b_f32(W, (wv * 3 + nl) * 16 + m15, ck, INTER);
#pragma unroll
                for (int m = 0; m < 4; ++m)
                    acc[m][nl] = wmma_bf16(af[m], b, acc[m][nl]);
            }
        }
    }
}

// ---------------------------------------------------------------- kernel 2: expert FFN
__global__ __launch_bounds__(FFN_THREADS)
void expert_ffn_kernel(const __bf16* __restrict__ xb,
                       const float* __restrict__ wg,
                       const float* __restrict__ wu,
                       const float* __restrict__ wd,
                       const int*   __restrict__ ecount,
                       const int*   __restrict__ etok,
                       const float* __restrict__ ewt,
                       float* __restrict__ out) {
    extern __shared__ char smem[];
    __bf16* Xs   = (__bf16*)smem;                        // [64][XS_STRIDE]
    __bf16* Gs   = (__bf16*)(smem + XS_BYTES);           // silu(g) -> h [64][HS_STRIDE]
    int*    tokS = (int*)  (smem + XS_BYTES + GS_BYTES);
    float*  wtsS = (float*)(smem + XS_BYTES + GS_BYTES + 64 * 4);

    const int e    = blockIdx.y;
    const int cnt  = ecount[e];
    const int base = blockIdx.x * MTILE;
    if (base >= cnt) return;                             // uniform exit

    const int tid  = threadIdx.x;
    const int lane = tid & 31;
    const int wv   = tid >> 5;                           // 0..15
    const int m15  = lane & 15;
    const int ko   = (lane >> 4) << 3;
    const int kb16 = (lane >> 4) << 4;
    const int rsub = (lane >> 4) << 3;                   // C-frag row sub-offset

    if (tid < MTILE) {
        int r = base + tid;
        if (r < cnt) { tokS[tid] = etok[(size_t)e * TOKENS + r];
                       wtsS[tid] = ewt [(size_t)e * TOKENS + r]; }
        else         { tokS[tid] = 0; wtsS[tid] = 0.0f; }    // pad rows: weight 0
    }
    __syncthreads();

    const float* wg_e = wg + (size_t)e * HID * INTER;
    const float* wu_e = wu + (size_t)e * HID * INTER;

    v8f acc[4][3];

    // ---- pass 1: gate GEMM, then silu(g) -> Gs (bf16) ----
#pragma unroll
    for (int m = 0; m < 4; ++m)
#pragma unroll
        for (int nl = 0; nl < 3; ++nl)
#pragma unroll
            for (int r = 0; r < 8; ++r) acc[m][nl][r] = 0.0f;
    gemm_xw(xb, wg_e, Xs, tokS, tid, lane, wv, acc);
#pragma unroll
    for (int m = 0; m < 4; ++m)
#pragma unroll
        for (int nl = 0; nl < 3; ++nl) {
            int nb = (wv * 3 + nl) * 16;
#pragma unroll
            for (int r = 0; r < 8; ++r) {
                float g = acc[m][nl][r];
                float sg = g / (1.0f + __expf(-g));
                Gs[(m * 16 + r + rsub) * HS_STRIDE + nb + m15] = (__bf16)sg;
            }
        }
    // no barrier needed: each thread re-reads exactly the Gs elements it wrote

    // ---- pass 2: up GEMM, then h = silu(g) * u -> Gs (bf16, in place) ----
#pragma unroll
    for (int m = 0; m < 4; ++m)
#pragma unroll
        for (int nl = 0; nl < 3; ++nl)
#pragma unroll
            for (int r = 0; r < 8; ++r) acc[m][nl][r] = 0.0f;
    gemm_xw(xb, wu_e, Xs, tokS, tid, lane, wv, acc);
#pragma unroll
    for (int m = 0; m < 4; ++m)
#pragma unroll
        for (int nl = 0; nl < 3; ++nl) {
            int nb = (wv * 3 + nl) * 16;
#pragma unroll
            for (int r = 0; r < 8; ++r) {
                int idx = (m * 16 + r + rsub) * HS_STRIDE + nb + m15;
                float h = (float)Gs[idx] * acc[m][nl][r];
                Gs[idx] = (__bf16)h;
            }
        }
    __syncthreads();                                     // h visible to all waves

    // ---- down-projection: [64 x 768] @ [768 x 2048], weighted atomic combine ----
    const float* wd_e = wd + (size_t)e * INTER * HID;
    for (int nt = wv * 8; nt < wv * 8 + 8; ++nt) {
        const int nb = nt * 16;
        v8f dacc[4];
#pragma unroll
        for (int m = 0; m < 4; ++m)
#pragma unroll
            for (int r = 0; r < 8; ++r) dacc[m][r] = 0.0f;

        for (int kk = 0; kk < INTER; kk += 32) {
            v16bf b = load_b_f32(wd_e, nb + m15, kk + kb16, HID);
#pragma unroll
            for (int m = 0; m < 4; ++m) {
                v16bf a = load_a_lds(Gs + (m * 16 + m15) * HS_STRIDE + kb16 + kk);
                dacc[m] = wmma_bf16(a, b, dacc[m]);
            }
        }
#pragma unroll
        for (int m = 0; m < 4; ++m)
#pragma unroll
            for (int r = 0; r < 8; ++r) {
                int rowm = m * 16 + r + rsub;
                int t    = tokS[rowm];
                float w  = wtsS[rowm];
                atomicAdd(out + (size_t)t * HID + nb + m15, w * dacc[m][r]);
            }
    }
}

// ---------------------------------------------------------------- launch
extern "C" void kernel_launch(void* const* d_in, const int* in_sizes, int n_in,
                              void* d_out, int out_size, void* d_ws, size_t ws_size,
                              hipStream_t stream) {
    const float* x   = (const float*)d_in[0];   // [T, H]
    const float* gw  = (const float*)d_in[1];   // [H, E]
    const float* wgp = (const float*)d_in[2];   // [E, H, I]
    const float* wup = (const float*)d_in[3];   // [E, H, I]
    const float* wdp = (const float*)d_in[4];   // [E, I, H]
    float* out = (float*)d_out;                 // [T, H]

    char* ws = (char*)d_ws;
    int*    ecount = (int*)ws;                                            // 256 B
    int*    etok   = (int*)  (ws + 256);                                  // 2 MB
    float*  ewt    = (float*)(ws + 256 + (size_t)NEXP * TOKENS * 4);      // 2 MB
    __bf16* xbf    = (__bf16*)(ws + 256 + 2 * (size_t)NEXP * TOKENS * 4); // 32 MB

    prep_kernel<<<8192, 256, 0, stream>>>(x, xbf, out, ecount, (long)TOKENS * HID);
    router_kernel<<<TOKENS / 8, 256, 0, stream>>>(x, gw, ecount, etok, ewt);
    expert_ffn_kernel<<<dim3(TOKENS / MTILE, NEXP), FFN_THREADS, SMEM_BYTES, stream>>>(
        xbf, wgp, wup, wdp, ecount, etok, ewt, out);
}
